// TemporalAttentionHorizonDecoder_11501922418698
// MI455X (gfx1250) — compile-verified, matrix operand order checked by reading, not run on hardware
//
#include <hip/hip_runtime.h>
#include <math.h>

// Problem constants (from reference)
#define B_   32
#define N_   128
#define C_   256
#define L_   96
#define H_   24
#define DH_  64
#define HID_ 1024
#define F_   320   // C + d_h

typedef __bf16 bf16;
typedef __attribute__((ext_vector_type(16))) __bf16 v16bf;
typedef __attribute__((ext_vector_type(8)))  __bf16 v8bf;
typedef __attribute__((ext_vector_type(8)))  float  v8f;

static __device__ __forceinline__ v8f wmma_bf16(v16bf a, v16bf b, v8f c) {
  // D = A(16x32 bf16) * B(32x16 bf16) + C(16x16 f32)
  return __builtin_amdgcn_wmma_f32_16x16x32_bf16(
      /*neg_a=*/false, a, /*neg_b=*/false, b,
      /*c_mod=*/(short)0, c, /*reuse_a=*/false, /*reuse_b=*/false);
}

// Fragment loader for 16-bit operands whose per-lane K-values are two
// contiguous 8-element runs in memory:
//   A matrix (MxK row-major):                r = m = lane&15
//   B matrix sourced from B^T (NxK r-major): r = n = lane&15
// ISA layout: lanes 0-15 hold K={0..7,16..23}; lanes 16-31 hold K={8..15,24..31}.
static __device__ __forceinline__ v16bf load_frag_rr(const bf16* src, int ld,
                                                     int row0, int k0, int lane) {
  int r  = lane & 15;
  int kb = (lane >> 4) << 3;
  const bf16* base = src + (row0 + r) * ld + k0 + kb;
  v8bf lo = *(const v8bf*)(base);
  v8bf hi = *(const v8bf*)(base + 16);
  v16bf out;
#pragma unroll
  for (int e = 0; e < 8; ++e) { out[e] = lo[e]; out[e + 8] = hi[e]; }
  return out;
}

static __device__ __forceinline__ v16bf load_frag_rr_bounded(const bf16* src, int ld,
                                                             int row0, int k0, int lane,
                                                             int rmax) {
  int r  = lane & 15;
  int kb = (lane >> 4) << 3;
  v16bf out;
  if (row0 + r < rmax) {
    const bf16* base = src + (row0 + r) * ld + k0 + kb;
    v8bf lo = *(const v8bf*)(base);
    v8bf hi = *(const v8bf*)(base + 16);
#pragma unroll
    for (int e = 0; e < 8; ++e) { out[e] = lo[e]; out[e + 8] = hi[e]; }
  } else {
#pragma unroll
    for (int e = 0; e < 16; ++e) out[e] = (bf16)0.0f;
  }
  return out;
}

// B fragment from a true row-major KxN source (column gather, scalar LDS reads).
static __device__ __forceinline__ v16bf load_bfrag_km(const bf16* src, int ldb,
                                                      int k0, int n0, int lane) {
  int ncol = lane & 15;
  int kb   = (lane >> 4) << 3;
  const bf16* col = src + n0 + ncol;
  v16bf out;
#pragma unroll
  for (int e = 0; e < 8; ++e) out[e]     = col[(k0 + kb + e)      * ldb];
#pragma unroll
  for (int e = 0; e < 8; ++e) out[8 + e] = col[(k0 + kb + 16 + e) * ldb];
  return out;
}

__global__ void f32_to_bf16_kernel(const float* __restrict__ src,
                                   unsigned short* __restrict__ dst, int n) {
  int i = blockIdx.x * blockDim.x + threadIdx.x;
  if (i < n) {
    bf16 v = (bf16)src[i];
    dst[i] = __builtin_bit_cast(unsigned short, v);
  }
}

__global__ __launch_bounds__(256)
void tahd_fused_kernel(const float* __restrict__ rep_seq,
                       const unsigned short* __restrict__ h_emb_bf,  // (24,64)
                       const unsigned short* __restrict__ key_w_bf,  // (64,256)
                       const unsigned short* __restrict__ w1_bf,     // (1024,320)
                       const float* __restrict__ b1,
                       const float* __restrict__ w2,
                       const float* __restrict__ b2,
                       float* __restrict__ out) {
  // LDS leading dims (elements); each row is a multiple of 16 bytes.
  constexpr int LDR = 264;  // rep   [96][264] bf16 = 50688 B
  constexpr int LDK = 72;   // k     [96][72]  bf16 = 13824 B (overlay lo)
  constexpr int LDS = 36;   // scoT  [96][36]  f32  = 13824 B (overlay hi)
  constexpr int LDA = 104;  // attn  [32][104] bf16 = 6656 B
  constexpr int LDX = 328;  // x     [32][328] bf16 = 20992 B (overlay, reused)

  __shared__ alignas(16) bf16 s_rep[L_ * LDR];
  __shared__ alignas(16) unsigned char s_ovl[27648];
  __shared__ alignas(16) bf16 s_attn[32 * LDA];
  __shared__ float s_y[32];

  bf16*  s_k  = (bf16*)s_ovl;
  float* s_sc = (float*)(s_ovl + 13824);
  bf16*  s_x  = (bf16*)s_ovl;   // valid after softmax barrier

  const int bn   = blockIdx.x;          // 0..4095
  const int bb   = bn >> 7;             // batch
  const int nn   = bn & 127;            // node
  const int tid  = threadIdx.x;
  const int lane = tid & 31;
  // Wave id is uniform across the wave: pin it into an SGPR so all tile loops
  // become scalar control flow (EXEC provably all-ones around every WMMA).
  const int wave = __builtin_amdgcn_readfirstlane(tid >> 5);

  // ---- Phase A: stage rep tile (transpose CxL -> LxC, f32 -> bf16) ----
  // L = 96 is a multiple of 4, so 4 consecutive elements never cross a row:
  // read float4 (global_load_b128), write 4 bf16 down one LDS column.
  {
    const float4* gsrc4 =
        (const float4*)(rep_seq + (size_t)bn * (C_ * L_));
    for (int q = tid; q < (C_ * L_) / 4; q += 256) {
      float4 v = gsrc4[q];
      int base = q * 4;
      int c = base / L_;
      int l = base - c * L_;
      bf16* dst = &s_rep[l * LDR + c];
      dst[0 * LDR] = (bf16)v.x;
      dst[1 * LDR] = (bf16)v.y;
      dst[2 * LDR] = (bf16)v.z;
      dst[3 * LDR] = (bf16)v.w;
    }
  }
  if (tid < 32) s_y[tid] = 0.0f;
  __syncthreads();

  // ---- G1: k = (rep @ key_w^T) * 1/sqrt(d_h);  M=96 N=64 K=256 ----
  // 24 tiles over 8 waves: exactly 3 per wave (scalar loop).
  for (int t = wave; t < 24; t += 8) {
    int mi = t >> 2, ni = t & 3;
    v8f acc = {0, 0, 0, 0, 0, 0, 0, 0};
#pragma unroll
    for (int kc = 0; kc < 8; ++kc) {
      v16bf a = load_frag_rr(s_rep, LDR, mi * 16, kc * 32, lane);
      v16bf b = load_frag_rr((const bf16*)key_w_bf, C_, ni * 16, kc * 32, lane);
      acc = wmma_bf16(a, b, acc);
    }
    int mb = (lane >> 4) << 3, nc = lane & 15;
#pragma unroll
    for (int i = 0; i < 8; ++i)
      s_k[(mi * 16 + mb + i) * LDK + ni * 16 + nc] = (bf16)(acc[i] * 0.125f);
  }
  __syncthreads();

  // ---- G2: scoresT (96 x 32pad) = k @ h_emb^T;  M=96 N=32 K=64 ----
  for (int t = wave; t < 12; t += 8) {
    int mi = t >> 1, ni = t & 1;
    v8f acc = {0, 0, 0, 0, 0, 0, 0, 0};
#pragma unroll
    for (int ks = 0; ks < 2; ++ks) {
      v16bf a = load_frag_rr(s_k, LDK, mi * 16, ks * 32, lane);
      v16bf b = load_frag_rr_bounded((const bf16*)h_emb_bf, DH_, ni * 16,
                                     ks * 32, lane, H_);
      acc = wmma_bf16(a, b, acc);
    }
    int mb = (lane >> 4) << 3, nc = lane & 15;
#pragma unroll
    for (int i = 0; i < 8; ++i)
      s_sc[(mi * 16 + mb + i) * LDS + ni * 16 + nc] = acc[i];
  }
  __syncthreads();

  // ---- G3: softmax over L per horizon h; attn stored bf16, rows 24..31 zero ----
  if (tid < 32) {
    int h = tid;
    if (h < H_) {
      float mx = -3.0e38f;
      for (int l = 0; l < L_; ++l) mx = fmaxf(mx, s_sc[l * LDS + h]);
      float sum = 0.0f;
      for (int l = 0; l < L_; ++l) {
        float e = __expf(s_sc[l * LDS + h] - mx);
        s_sc[l * LDS + h] = e;
        sum += e;
      }
      float inv = 1.0f / sum;
      for (int l = 0; l < L_; ++l)
        s_attn[h * LDA + l] = (bf16)(s_sc[l * LDS + h] * inv);
    } else {
      for (int l = 0; l < L_; ++l) s_attn[h * LDA + l] = (bf16)0.0f;
    }
  }
  __syncthreads();

  // ---- q fill: x[:, 256:320] = h_emb (rows >= 24 zero). Overlays s_k/s_sc. ----
  for (int idx = tid; idx < 32 * DH_; idx += 256) {
    int r = idx >> 6, d = idx & 63;
    s_x[r * LDX + C_ + d] =
        (r < H_) ? ((const bf16*)h_emb_bf)[r * DH_ + d] : (bf16)0.0f;
  }
  __syncthreads();

  const int mt = wave & 1;   // m-tile (rows 0..15 / 16..31), scalar
  const int wg = wave >> 1;  // 0..3, scalar

  // ---- G4: x[:, :256] = attn @ rep;  M=32 N=256 K=96 ----
  for (int j = 0; j < 4; ++j) {
    int nt = wg * 4 + j;
    v8f acc = {0, 0, 0, 0, 0, 0, 0, 0};
#pragma unroll
    for (int ks = 0; ks < 3; ++ks) {
      v16bf a = load_frag_rr(s_attn, LDA, mt * 16, ks * 32, lane);
      v16bf b = load_bfrag_km(s_rep, LDR, ks * 32, nt * 16, lane);
      acc = wmma_bf16(a, b, acc);
    }
    int mb = (lane >> 4) << 3, nc = lane & 15;
#pragma unroll
    for (int i = 0; i < 8; ++i)
      s_x[(mt * 16 + mb + i) * LDX + nt * 16 + nc] = (bf16)acc[i];
  }
  __syncthreads();

  // ---- G5: MLP layer1 (M=32 N=1024 K=320) + bias + exact GELU + fused dot w2 ----
  v16bf afr[10];
#pragma unroll
  for (int ks = 0; ks < 10; ++ks)
    afr[ks] = load_frag_rr(s_x, LDX, mt * 16, ks * 32, lane);

  float lsum[8] = {0.f, 0.f, 0.f, 0.f, 0.f, 0.f, 0.f, 0.f};
  const int ncol = lane & 15;
  for (int j = 0; j < 16; ++j) {
    int nt = wg * 16 + j;
    int o  = nt * 16 + ncol;
    v8f acc = {0, 0, 0, 0, 0, 0, 0, 0};
#pragma unroll
    for (int ks = 0; ks < 10; ++ks) {
      v16bf b = load_frag_rr((const bf16*)w1_bf, F_, nt * 16, ks * 32, lane);
      acc = wmma_bf16(afr[ks], b, acc);
    }
    float bias = b1[o];
    float wv   = w2[o];
#pragma unroll
    for (int i = 0; i < 8; ++i) {
      float v  = acc[i] + bias;
      float ge = 0.5f * v * (1.0f + erff(v * 0.70710678118654752f));
      lsum[i] += ge * wv;
    }
  }
  {
    int mb = (lane >> 4) << 3;
#pragma unroll
    for (int i = 0; i < 8; ++i)
      atomicAdd(&s_y[mt * 16 + mb + i], lsum[i]);
  }
  __syncthreads();

  // ---- output: y[b,h,n,0] ----
  if (tid < H_) {
    out[((size_t)bb * H_ + tid) * N_ + nn] = s_y[tid] + b2[0];
  }
}

extern "C" void kernel_launch(void* const* d_in, const int* in_sizes, int n_in,
                              void* d_out, int out_size, void* d_ws, size_t ws_size,
                              hipStream_t stream) {
  const float* rep_seq = (const float*)d_in[0];
  const float* h_emb   = (const float*)d_in[1];
  const float* key_w   = (const float*)d_in[2];
  const float* w1      = (const float*)d_in[3];
  const float* b1      = (const float*)d_in[4];
  const float* w2      = (const float*)d_in[5];
  const float* b2      = (const float*)d_in[6];

  // Workspace layout (bf16 = unsigned short elements)
  unsigned short* ws        = (unsigned short*)d_ws;
  unsigned short* key_w_bf  = ws;                     // 64*256   = 16384
  unsigned short* h_emb_bf  = ws + 16384;             // 24*64    = 1536
  unsigned short* w1_bf     = ws + 16384 + 1536;      // 1024*320 = 327680

  const int n_kw = DH_ * C_;
  const int n_he = H_ * DH_;
  const int n_w1 = HID_ * F_;
  f32_to_bf16_kernel<<<(n_kw + 255) / 256, 256, 0, stream>>>(key_w, key_w_bf, n_kw);
  f32_to_bf16_kernel<<<(n_he + 255) / 256, 256, 0, stream>>>(h_emb, h_emb_bf, n_he);
  f32_to_bf16_kernel<<<(n_w1 + 255) / 256, 256, 0, stream>>>(w1, w1_bf, n_w1);

  tahd_fused_kernel<<<B_ * N_, 256, 0, stream>>>(
      rep_seq, h_emb_bf, key_w_bf, w1_bf, b1, w2, b2, (float*)d_out);
}
